// SpatioTemporalGraphModel_7258494730708
// MI455X (gfx1250) — matrix operand order repaired and checked by reading, not compiled, not run on hardware
//
#include <hip/hip_runtime.h>
#include <hip/hip_bf16.h>
#include <math.h>

// ---------------- model constants ----------------
#define Bx    16
#define Nn    2000
#define Wt    32
#define Ff    5
#define THh   128
#define SHh   64
#define HEADS 4
#define Mn    (Bx * Nn)          // 32000 rows
#define EB    32000              // base edges per graph
#define ETOT  (Bx * EB + Mn)     // batched edges + self loops = 544000
#define NT3   24                 // 384/16 column tiles of the 3-gate matmuls
#define KC4   4                  // 128/32 K chunks

// packed-weight region sizes (in _Float16 elements)
#define PK_MAT   (KC4 * NT3 * 32 * 16)   // 49152 halfs = 96KB (one 128x384 B matrix)
#define PK_GATW  (KC4 * 4   * 32 * 16)   // 8192 halfs        (128x64 B matrix)
#define PK_XW    (NT3 * 32 * 16)         // 12288 halfs       (zero-padded 32x384 Wih0 chunk)
#define PK_TOTAL (3 * PK_MAT + PK_GATW + PK_XW)   // 167936 halfs = 335872 B

// workspace layout (bytes; PK_TOTAL*2 is 256-aligned)
#define OFF_PK     0
#define OFF_TEMB   335872
#define OFF_XP     (OFF_TEMB + Mn * THh * 4)
#define OFF_ASRC   (OFF_XP   + Mn * SHh * 4)
#define OFF_ADST   (OFF_ASRC + Mn * HEADS * 4)
#define OFF_ALPHA  (OFF_ADST + Mn * HEADS * 4)
#define OFF_AMAX   (OFF_ALPHA + (size_t)ETOT * HEADS * 4)
#define OFF_DEN    (OFF_AMAX + Mn * HEADS * 4)
#define OFF_AGG    (OFF_DEN  + Mn * HEADS * 4)
// total ~ 44 MB of workspace

typedef __attribute__((ext_vector_type(16))) _Float16 v16h;
typedef __attribute__((ext_vector_type(8)))  float    v8f;
typedef int av4i __attribute__((vector_size(16)));   // matches async builtin param

__device__ __forceinline__ v8f wmma_f16(v16h a, v16h b, v8f c) {
  return __builtin_amdgcn_wmma_f32_16x16x32_f16(false, a, false, b, (short)0, c,
                                                false, false);
}

// ---- fast transcendentals (v_exp_f32 / v_rcp_f32 / v_tanh_f32) ----
__device__ __forceinline__ float fast_sigmoid(float x) {
  return __builtin_amdgcn_rcpf(1.f + __builtin_amdgcn_exp2f(-1.442695041f * x));
}
__device__ __forceinline__ float fast_tanh(float x) {
#if defined(__has_builtin) && __has_builtin(__builtin_amdgcn_tanhf)
  return __builtin_amdgcn_tanhf(x);
#else
  float e = __builtin_amdgcn_exp2f(2.885390082f * x);   // exp(2x)
  return 1.f - 2.f * __builtin_amdgcn_rcpf(e + 1.f);
#endif
}

// ---- CDNA5 async global->LDS staging (guarded; sync fallback) ----
#if defined(__has_builtin)
#if __has_builtin(__builtin_amdgcn_global_load_async_to_lds_b128) && \
    __has_builtin(__builtin_amdgcn_s_wait_asynccnt)
#define USE_ASYNC_LDS 1
#endif
#endif

__device__ __forceinline__ void stage_b128(_Float16* dst, const _Float16* src) {
#ifdef USE_ASYNC_LDS
  __builtin_amdgcn_global_load_async_to_lds_b128(
      (__attribute__((address_space(1))) av4i*)(_Float16*)src,
      (__attribute__((address_space(3))) av4i*)dst, 0, 0);
#else
  *(int4*)dst = *(const int4*)src;
#endif
}
__device__ __forceinline__ void stage_fence() {
#ifdef USE_ASYNC_LDS
  __builtin_amdgcn_s_wait_asynccnt(0);
#endif
}

// order-preserving float -> uint key for atomic segment-max
__device__ __forceinline__ unsigned fkey(float f) {
  unsigned u = __float_as_uint(f);
  return (u & 0x80000000u) ? ~u : (u | 0x80000000u);
}
__device__ __forceinline__ float finv(unsigned k) {
  unsigned u = (k & 0x80000000u) ? (k & 0x7FFFFFFFu) : ~k;
  return __uint_as_float(u);
}

// A-fragment (16-bit A 16x32) address helper: element (row r, k') of a tile
// lives at lane (r + 16*hiA), half index 2*j + e inside the 16-half lane chunk.
__device__ __forceinline__ int afrag_off(int rloc, int kp) {
  int hiA = (kp >> 3) & 1;
  int jA  = ((kp & 7) >> 1) + 4 * (kp >> 4);
  int eA  = kp & 1;
  return (rloc + 16 * hiA) * 16 + 2 * jA + eA;
}

// ---------------------------------------------------------------------------
// Pack weights into per-lane WMMA B-fragment layout (f16):
//   pk[kc][nt][lane][16],  b[2j+e] = B[kc*32 + 16*(j>>2) + 8*(lane>>4) + 2*(j&3) + e][nt*16 + (lane&15)]
//   B[k][n] = W[n][k] for the GRU mats; gat_W is stored [K,N] -> direct.
//   Region 4: zero-padded K=32 chunk holding Wih0 (only k<5 nonzero).
// ---------------------------------------------------------------------------
__global__ void pack_weights_kernel(const float* __restrict__ whh0,
                                    const float* __restrict__ wih1,
                                    const float* __restrict__ whh1,
                                    const float* __restrict__ gatw,
                                    const float* __restrict__ wih0,
                                    _Float16* __restrict__ pk) {
  int idx = blockIdx.x * blockDim.x + threadIdx.x;
  if (idx >= PK_TOTAL) return;
  int halfidx = idx & 15;
  int lane    = (idx >> 4) & 31;
  int n  = lane & 15, hi = lane >> 4;
  int j  = halfidx >> 1, e = halfidx & 1;
  float v;
  if (idx < 3 * PK_MAT) {
    int mat = idx / PK_MAT;
    int tile = (idx - mat * PK_MAT) >> 9;            // kc*NT3 + nt
    int kc = tile / NT3, nt = tile - kc * NT3;
    int k = kc * 32 + 16 * (j >> 2) + 8 * hi + 2 * (j & 3) + e;
    int nout = nt * 16 + n;
    const float* src = (mat == 0) ? whh0 : (mat == 1) ? wih1 : whh1;
    v = src[nout * THh + k];
  } else if (idx < 3 * PK_MAT + PK_GATW) {
    int tile = (idx - 3 * PK_MAT) >> 9;              // kc*4 + h
    int kc = tile >> 2, nt = tile & 3;
    int k = kc * 32 + 16 * (j >> 2) + 8 * hi + 2 * (j & 3) + e;
    v = gatw[k * SHh + nt * 16 + n];                 // gat_W is [TH, H*C]
  } else {
    int nt = (idx - 3 * PK_MAT - PK_GATW) >> 9;      // single K chunk
    int k = 16 * (j >> 2) + 8 * hi + 2 * (j & 3) + e;
    v = (k < Ff) ? wih0[(nt * 16 + n) * Ff + k] : 0.f;
  }
  pk[idx] = (_Float16)v;
}

// ---------------------------------------------------------------------------
// Fused 2-layer GRU over 32 steps + LayerNorm -> temb [M,128].
// 128 rows / workgroup, 8 waves, wave owns a 16-row slice.
// LDS: bufA (streamed Whh0 / Wih1), resident Whh1 + Wih0-chunk, h-states in
// A-fragment layout, biases. x enters layer-0 through an augmented-K WMMA.
// ---------------------------------------------------------------------------
__global__ void __launch_bounds__(256) gru_kernel(
    const float* __restrict__ x,
    const float* __restrict__ b_ih0, const float* __restrict__ b_hh0,
    const float* __restrict__ b_ih1, const float* __restrict__ b_hh1,
    const float* __restrict__ ln1g,  const float* __restrict__ ln1b,
    const _Float16* __restrict__ pk,
    float* __restrict__ temb) {
  extern __shared__ char smem_raw[];
  _Float16* bufA  = (_Float16*)smem_raw;         // 49152 halfs: Whh0 then Wih1
  _Float16* bufHH1= bufA + PK_MAT;               // 49152 halfs: resident Whh1
  _Float16* bufX0 = bufHH1 + PK_MAT;             // 12288 halfs: resident Wih0 chunk
  _Float16* h0    = bufX0 + PK_XW;               // 16384 halfs
  _Float16* h1    = h0 + 8 * 2048;               // 16384 halfs
  float* bi0 = (float*)(h1 + 8 * 2048);          // 4 x 384 f32 biases
  float* bh0 = bi0 + 384;
  float* bi1 = bh0 + 384;
  float* bh1 = bi1 + 384;

  const int tid  = threadIdx.x;
  const int lane = tid & 31;
  const int wv   = tid >> 5;
  const int nCol = lane & 15;
  const int hiAcc = lane >> 4;
  const int blockRow = blockIdx.x * 128;

  const _Float16* pkWhh0 = pk;
  const _Float16* pkWih1 = pk + PK_MAT;
  const _Float16* pkWhh1 = pk + 2 * PK_MAT;
  const _Float16* pkXW   = pk + 3 * PK_MAT + PK_GATW;

  // one-time staging: resident Whh1, Wih0 chunk, biases; zero h states
  for (int i = tid; i < PK_MAT / 8; i += 256) stage_b128(bufHH1 + i * 8, pkWhh1 + i * 8);
  for (int i = tid; i < PK_XW / 8; i += 256)  stage_b128(bufX0 + i * 8, pkXW + i * 8);
  for (int i = tid; i < 384; i += 256) {
    bi0[i] = b_ih0[i]; bh0[i] = b_hh0[i]; bi1[i] = b_ih1[i]; bh1[i] = b_hh1[i];
  }
  _Float16* h0s = h0 + wv * 2048;   // per-wave [kc][32][16]
  _Float16* h1s = h1 + wv * 2048;
  for (int i = lane; i < 2048; i += 32) { h0s[i] = (_Float16)0.f; h1s[i] = (_Float16)0.f; }
  stage_fence();
  __syncthreads();

  for (int t = 0; t < Wt; ++t) {
    // ---- stage W_hh0 -> bufA (async b128 -> LDS) ----
    for (int i = tid; i < PK_MAT / 8; i += 256) stage_b128(bufA + i * 8, pkWhh0 + i * 8);
    stage_fence();
    __syncthreads();
    // warm L2/near caches for the next staged matrix while we compute
    __builtin_prefetch(pkWih1 + (size_t)tid * 192, 0, 1);

    // ---------------- layer 0 ----------------
    // x enters as an extra K=32 A-chunk: lanes 0-15 hold the 5 features
    v16h ax = {};
    if (hiAcc == 0) {
      const float* xr = x + ((size_t)(blockRow + wv * 16 + nCol) * Wt + t) * Ff;
      ax[0] = (_Float16)xr[0]; ax[1] = (_Float16)xr[1]; ax[2] = (_Float16)xr[2];
      ax[3] = (_Float16)xr[3]; ax[4] = (_Float16)xr[4];
    }
    v16h a0[KC4];
    #pragma unroll
    for (int kc = 0; kc < KC4; ++kc)
      a0[kc] = *(const v16h*)(h0s + (kc * 32 + lane) * 16);

    #pragma unroll 1
    for (int ct = 0; ct < 8; ++ct) {
      v8f accR = {}, accZ = {}, accHN = {}, accIN = {};
      #pragma unroll
      for (int kc = 0; kc < KC4; ++kc) {
        v16h bR = *(const v16h*)(bufA + ((kc * NT3 + ct     ) * 32 + lane) * 16);
        v16h bZ = *(const v16h*)(bufA + ((kc * NT3 + ct + 8 ) * 32 + lane) * 16);
        v16h bN = *(const v16h*)(bufA + ((kc * NT3 + ct + 16) * 32 + lane) * 16);
        accR  = wmma_f16(a0[kc], bR, accR);
        accZ  = wmma_f16(a0[kc], bZ, accZ);
        accHN = wmma_f16(a0[kc], bN, accHN);
      }
      {  // x-side contributions (augmented K chunk)
        v16h xR = *(const v16h*)(bufX0 + ((ct     ) * 32 + lane) * 16);
        v16h xZ = *(const v16h*)(bufX0 + ((ct + 8 ) * 32 + lane) * 16);
        v16h xN = *(const v16h*)(bufX0 + ((ct + 16) * 32 + lane) * 16);
        accR  = wmma_f16(ax, xR, accR);
        accZ  = wmma_f16(ax, xZ, accZ);
        accIN = wmma_f16(ax, xN, accIN);    // i_n kept separate from h_n
      }
      int c = ct * 16 + nCol;
      int kc_c = c >> 5, kp = c & 31;
      float br = bi0[c] + bh0[c];
      float bz = bi0[128 + c] + bh0[128 + c];
      float bin_ = bi0[256 + c], bhn_ = bh0[256 + c];
      #pragma unroll
      for (int v = 0; v < 8; ++v) {
        int rloc = v + 8 * hiAcc;
        _Float16* hp = h0s + kc_c * 512 + afrag_off(rloc, kp);
        float hold = (float)(*hp);
        float r = fast_sigmoid(accR[v] + br);
        float z = fast_sigmoid(accZ[v] + bz);
        float n = fast_tanh(accIN[v] + bin_ + r * (accHN[v] + bhn_));
        *hp = (_Float16)((1.f - z) * n + z * hold);
      }
    }
    __syncthreads();   // all waves done reading bufA (Whh0)

    // ---- stage W_ih1 -> bufA ----
    for (int i = tid; i < PK_MAT / 8; i += 256) stage_b128(bufA + i * 8, pkWih1 + i * 8);
    stage_fence();
    __syncthreads();
    __builtin_prefetch(pkWhh0 + (size_t)tid * 192, 0, 1);

    // ---------------- layer 1 ----------------
    v16h a0n[KC4], a1[KC4];
    #pragma unroll
    for (int kc = 0; kc < KC4; ++kc) {
      a0n[kc] = *(const v16h*)(h0s + (kc * 32 + lane) * 16);  // h0 (new)
      a1[kc]  = *(const v16h*)(h1s + (kc * 32 + lane) * 16);  // h1 (old)
    }
    #pragma unroll 1
    for (int ct = 0; ct < 8; ++ct) {
      v8f accR = {}, accZ = {}, accIN = {}, accHN = {};
      #pragma unroll
      for (int kc = 0; kc < KC4; ++kc) {
        v16h biR = *(const v16h*)(bufA   + ((kc * NT3 + ct     ) * 32 + lane) * 16);
        v16h biZ = *(const v16h*)(bufA   + ((kc * NT3 + ct + 8 ) * 32 + lane) * 16);
        v16h biN = *(const v16h*)(bufA   + ((kc * NT3 + ct + 16) * 32 + lane) * 16);
        v16h bhR = *(const v16h*)(bufHH1 + ((kc * NT3 + ct     ) * 32 + lane) * 16);
        v16h bhZ = *(const v16h*)(bufHH1 + ((kc * NT3 + ct + 8 ) * 32 + lane) * 16);
        v16h bhN = *(const v16h*)(bufHH1 + ((kc * NT3 + ct + 16) * 32 + lane) * 16);
        accR  = wmma_f16(a0n[kc], biR, accR);
        accR  = wmma_f16(a1[kc],  bhR, accR);
        accZ  = wmma_f16(a0n[kc], biZ, accZ);
        accZ  = wmma_f16(a1[kc],  bhZ, accZ);
        accIN = wmma_f16(a0n[kc], biN, accIN);
        accHN = wmma_f16(a1[kc],  bhN, accHN);
      }
      int c = ct * 16 + nCol;
      int kc_c = c >> 5, kp = c & 31;
      float br = bi1[c] + bh1[c];
      float bz = bi1[128 + c] + bh1[128 + c];
      float bin_ = bi1[256 + c], bhn_ = bh1[256 + c];
      #pragma unroll
      for (int v = 0; v < 8; ++v) {
        int rloc = v + 8 * hiAcc;
        _Float16* hp = h1s + kc_c * 512 + afrag_off(rloc, kp);
        float hold = (float)(*hp);
        float r = fast_sigmoid(accR[v] + br);
        float z = fast_sigmoid(accZ[v] + bz);
        float n = fast_tanh(accIN[v] + bin_ + r * (accHN[v] + bhn_));
        *hp = (_Float16)((1.f - z) * n + z * hold);
      }
    }
    __syncthreads();   // bufA restaged next step
  }

  // ---- LayerNorm(h1[-1]) -> temb  (lane pair (l, l^16) shares a row) ----
  {
    int r = lane & 15;
    int half = lane >> 4;
    float vals[64];
    float s = 0.f;
    #pragma unroll
    for (int i = 0; i < 64; ++i) {
      int c = half * 64 + i;
      float hv = (float)h1s[(c >> 5) * 512 + afrag_off(r, c & 31)];
      vals[i] = hv; s += hv;
    }
    float mean = (s + __shfl_xor(s, 16, 32)) * (1.f / 128.f);
    float q = 0.f;
    #pragma unroll
    for (int i = 0; i < 64; ++i) { float d = vals[i] - mean; q += d * d; }
    float rstd = rsqrtf((q + __shfl_xor(q, 16, 32)) * (1.f / 128.f) + 1e-5f);
    float* to = temb + (size_t)(blockRow + wv * 16 + r) * THh;
    #pragma unroll
    for (int i = 0; i < 64; ++i) {
      int c = half * 64 + i;
      to[c] = (vals[i] - mean) * rstd * ln1g[c] + ln1b[c];
    }
  }
}

// ---------------------------------------------------------------------------
// GAT projection: xp = temb @ gat_W via WMMA + per-head attention dots
// ---------------------------------------------------------------------------
__global__ void __launch_bounds__(256) gatproj_kernel(
    const float* __restrict__ temb, const _Float16* __restrict__ pkgatw,
    const float* __restrict__ att_src, const float* __restrict__ att_dst,
    float* __restrict__ xp, float* __restrict__ asrc, float* __restrict__ adst) {
  const int tid = threadIdx.x, lane = tid & 31, wv = tid >> 5;
  const int blockRow = blockIdx.x * 128;
  const int n = lane & 15, hi = lane >> 4;
  const int row = blockRow + wv * 16 + n;

  v16h a[KC4];
  #pragma unroll
  for (int kc = 0; kc < KC4; ++kc) {
    const float* tr = temb + (size_t)row * THh + kc * 32 + 8 * hi;
    v16h av;
    #pragma unroll
    for (int j = 0; j < 8; ++j) {
      int kb = (j < 4) ? 2 * j : 8 + 2 * j;
      float2 p = *(const float2*)(tr + kb);
      av[2 * j] = (_Float16)p.x; av[2 * j + 1] = (_Float16)p.y;
    }
    a[kc] = av;
  }
  #pragma unroll
  for (int h = 0; h < HEADS; ++h) {
    v8f acc = {};
    #pragma unroll
    for (int kc = 0; kc < KC4; ++kc) {
      v16h b = *(const v16h*)(pkgatw + ((kc * 4 + h) * 32 + lane) * 16);
      acc = wmma_f16(a[kc], b, acc);
    }
    float as = att_src[h * 16 + n], ad = att_dst[h * 16 + n];
    #pragma unroll
    for (int v = 0; v < 8; ++v) {
      int m = blockRow + wv * 16 + v + 8 * hi;
      xp[(size_t)m * SHh + h * 16 + n] = acc[v];
      float s1 = acc[v] * as, s2 = acc[v] * ad;
      #pragma unroll
      for (int off = 8; off >= 1; off >>= 1) {
        s1 += __shfl_xor(s1, off, 32);
        s2 += __shfl_xor(s2, off, 32);
      }
      if (n == 0) { asrc[m * HEADS + h] = s1; adst[m * HEADS + h] = s2; }
    }
  }
}

// ---------------------------------------------------------------------------
__global__ void init_kernel(unsigned* amax, float* den, float* agg) {
  int i = blockIdx.x * blockDim.x + threadIdx.x;
  if (i < Mn * HEADS) { amax[i] = 0u; den[i] = 0.f; }
  if (i < Mn * SHh) agg[i] = 0.f;
}

__device__ __forceinline__ void edge_decode(int e, const long long* ei, int& s, int& d) {
  if (e < Bx * EB) {
    int b = e / EB, k = e - b * EB;
    s = (int)ei[k] + b * Nn;
    d = (int)ei[EB + k] + b * Nn;
  } else {
    s = d = e - Bx * EB;   // self loops
  }
}

__global__ void edge_alpha_kernel(const long long* __restrict__ ei,
                                  const float* __restrict__ asrc,
                                  const float* __restrict__ adst,
                                  float* __restrict__ alpha, unsigned* __restrict__ amax) {
  int e = blockIdx.x * blockDim.x + threadIdx.x;
  if (e >= ETOT) return;
  int s, d; edge_decode(e, ei, s, d);
  #pragma unroll
  for (int h = 0; h < HEADS; ++h) {
    float al = asrc[s * HEADS + h] + adst[d * HEADS + h];
    al = (al > 0.f) ? al : 0.2f * al;        // leaky_relu 0.2
    alpha[(size_t)e * HEADS + h] = al;
    atomicMax(&amax[d * HEADS + h], fkey(al));
  }
}

__global__ void edge_exp_kernel(const long long* __restrict__ ei,
                                float* __restrict__ alpha,
                                const unsigned* __restrict__ amax,
                                float* __restrict__ den) {
  int e = blockIdx.x * blockDim.x + threadIdx.x;
  if (e >= ETOT) return;
  int s, d; edge_decode(e, ei, s, d);
  #pragma unroll
  for (int h = 0; h < HEADS; ++h) {
    float ex = __builtin_amdgcn_exp2f(1.442695041f *
               (alpha[(size_t)e * HEADS + h] - finv(amax[d * HEADS + h])));
    alpha[(size_t)e * HEADS + h] = ex;
    atomicAdd(&den[d * HEADS + h], ex);
  }
}

__global__ void edge_agg_kernel(const long long* __restrict__ ei,
                                const float* __restrict__ alpha,
                                const float* __restrict__ den,
                                const float* __restrict__ xp,
                                float* __restrict__ agg) {
  int e = blockIdx.x * blockDim.x + threadIdx.x;
  if (e >= ETOT) return;
  int s, d; edge_decode(e, ei, s, d);
  #pragma unroll
  for (int h = 0; h < HEADS; ++h) {
    float w = alpha[(size_t)e * HEADS + h] / den[d * HEADS + h];
    const float* xs = xp + (size_t)s * SHh + h * 16;
    float* ag = agg + (size_t)d * SHh + h * 16;
    #pragma unroll
    for (int c = 0; c < 16; ++c) atomicAdd(&ag[c], w * xs[c]);
  }
}

// ---------------------------------------------------------------------------
// Per-node: +gat_b, LayerNorm(64), leaky, 64->32 leaky, 32->1 -> out [B,N]
// ---------------------------------------------------------------------------
__global__ void head_kernel(const float* __restrict__ agg, const float* __restrict__ gat_b,
                            const float* __restrict__ g2, const float* __restrict__ b2,
                            const float* __restrict__ W1, const float* __restrict__ rb1,
                            const float* __restrict__ W2, const float* __restrict__ rb2,
                            float* __restrict__ out) {
  int m = blockIdx.x * blockDim.x + threadIdx.x;
  if (m >= Mn) return;
  float v[SHh];
  float s = 0.f;
  #pragma unroll
  for (int c = 0; c < SHh; ++c) { v[c] = agg[(size_t)m * SHh + c] + gat_b[c]; s += v[c]; }
  float mean = s * (1.f / SHh);
  float q = 0.f;
  #pragma unroll
  for (int c = 0; c < SHh; ++c) { float d = v[c] - mean; q += d * d; }
  float rstd = rsqrtf(q * (1.f / SHh) + 1e-5f);
  #pragma unroll
  for (int c = 0; c < SHh; ++c) {
    float y = (v[c] - mean) * rstd * g2[c] + b2[c];
    v[c] = (y > 0.f) ? y : 0.2f * y;
  }
  float p = rb2[0];
  for (int j = 0; j < 32; ++j) {
    float hj = rb1[j];
    #pragma unroll
    for (int c = 0; c < SHh; ++c) hj += v[c] * W1[c * 32 + j];
    hj = (hj > 0.f) ? hj : 0.2f * hj;
    p += hj * W2[j];
  }
  out[m] = p;
}

// ---------------------------------------------------------------------------
extern "C" void kernel_launch(void* const* d_in, const int* in_sizes, int n_in,
                              void* d_out, int out_size, void* d_ws, size_t ws_size,
                              hipStream_t stream) {
  const float*     x    = (const float*)d_in[0];
  const long long* ei   = (const long long*)d_in[1];
  const float* Wih0 = (const float*)d_in[2];
  const float* Whh0 = (const float*)d_in[3];
  const float* bih0 = (const float*)d_in[4];
  const float* bhh0 = (const float*)d_in[5];
  const float* Wih1 = (const float*)d_in[6];
  const float* Whh1 = (const float*)d_in[7];
  const float* bih1 = (const float*)d_in[8];
  const float* bhh1 = (const float*)d_in[9];
  const float* ln1g = (const float*)d_in[10];
  const float* ln1b = (const float*)d_in[11];
  const float* gatW = (const float*)d_in[12];
  const float* attS = (const float*)d_in[13];
  const float* attD = (const float*)d_in[14];
  const float* gatb = (const float*)d_in[15];
  const float* ln2g = (const float*)d_in[16];
  const float* ln2b = (const float*)d_in[17];
  const float* W1   = (const float*)d_in[18];
  const float* rb1  = (const float*)d_in[19];
  const float* W2   = (const float*)d_in[20];
  const float* rb2  = (const float*)d_in[21];

  char* ws = (char*)d_ws;
  _Float16* pk   = (_Float16*)(ws + OFF_PK);
  float* temb    = (float*)(ws + OFF_TEMB);
  float* xp      = (float*)(ws + OFF_XP);
  float* asrc    = (float*)(ws + OFF_ASRC);
  float* adst    = (float*)(ws + OFF_ADST);
  float* alpha   = (float*)(ws + OFF_ALPHA);
  unsigned* amax = (unsigned*)(ws + OFF_AMAX);
  float* den     = (float*)(ws + OFF_DEN);
  float* agg     = (float*)(ws + OFF_AGG);

  pack_weights_kernel<<<(PK_TOTAL + 255) / 256, 256, 0, stream>>>(Whh0, Wih1, Whh1,
                                                                  gatW, Wih0, pk);

  // dynamic LDS: streamed 96KB + resident 96KB + 24KB + 2x32KB h + biases
  const size_t smem = (size_t)(2 * PK_MAT + PK_XW + 2 * 8 * 2048) * sizeof(_Float16) +
                      (size_t)(4 * 384) * sizeof(float);
  gru_kernel<<<Mn / 128, 256, smem, stream>>>(x, bih0, bhh0, bih1, bhh1,
                                              ln1g, ln1b, pk, temb);

  init_kernel<<<(Mn * SHh) / 256, 256, 0, stream>>>(amax, den, agg);
  gatproj_kernel<<<Mn / 128, 256, 0, stream>>>(temb, pk + 3 * PK_MAT, attS, attD,
                                               xp, asrc, adst);
  edge_alpha_kernel<<<(ETOT + 255) / 256, 256, 0, stream>>>(ei, asrc, adst, alpha, amax);
  edge_exp_kernel<<<(ETOT + 255) / 256, 256, 0, stream>>>(ei, alpha, amax, den);
  edge_agg_kernel<<<(ETOT + 255) / 256, 256, 0, stream>>>(ei, alpha, den, xp, agg);
  head_kernel<<<(Mn + 255) / 256, 256, 0, stream>>>(agg, gatb, ln2g, ln2b,
                                                    W1, rb1, W2, rb2, (float*)d_out);
}
